// LimitedContextAttention_82257213653605
// MI455X (gfx1250) — matrix operand order repaired and checked by reading, not compiled, not run on hardware
//
#include <hip/hip_runtime.h>
#include <hip/hip_bf16.h>

typedef __bf16 bf16_t;
typedef __attribute__((ext_vector_type(16))) bf16_t v16bf;
typedef __attribute__((ext_vector_type(8)))  bf16_t v8bf;
typedef __attribute__((ext_vector_type(8)))  float  v8f;

union Frag16 { v16bf v; v8bf h[2]; };

#define WMMA_BF16(a, b, c) \
  __builtin_amdgcn_wmma_f32_16x16x32_bf16(false, (a), false, (b), (short)0, (c), false, false)

// ---------------- CDNA5 async global->LDS copy (guarded) ----------------
#if defined(__HIP_DEVICE_COMPILE__) && \
    __has_builtin(__builtin_amdgcn_global_load_async_to_lds_b128)
#define HAS_ASYNC_LDS 1
typedef int v4i_g __attribute__((vector_size(16)));   // matches builtin's V4i pointee
__device__ __forceinline__ void stage16(const bf16_t* g, bf16_t* l) {
  // one async b128 per lane: LDS written by the async unit, tracked by ASYNCcnt
  __builtin_amdgcn_global_load_async_to_lds_b128(
      (__attribute__((address_space(1))) v4i_g*)g,
      (__attribute__((address_space(3))) v4i_g*)l, 0, 0);
}
#if __has_builtin(__builtin_amdgcn_s_wait_asynccnt)
#define ASYNC_WAIT(n) __builtin_amdgcn_s_wait_asynccnt(n)
#else
#define ASYNC_WAIT(n) asm volatile("s_wait_asynccnt %0" ::"i"(n) : "memory")
#endif
#else
#define HAS_ASYNC_LDS 0
__device__ __forceinline__ void stage16(const bf16_t* g, bf16_t* l) {
  *(uint4*)l = *(const uint4*)g;
}
#define ASYNC_WAIT(n) do {} while (0)
#endif

__device__ __forceinline__ void stage_tile32(const bf16_t* g, bf16_t* l) {
  stage16(g, l);
  stage16(g + 8, l + 8);
}

// ---------------- problem dimensions ----------------
constexpr int D   = 1024;
constexpr int NH  = 16;
constexpr int HD  = 64;
constexpr int S   = 4096;
constexpr int NB  = 4;          // batch
constexpr int C   = 128;        // window block
constexpr int M   = NB * S;     // 16384 rows
constexpr int SP  = S + 2 * C;  // 4352 padded seq
constexpr int WIN = 3 * C;      // 384 window

// ---------------- workspace layout (bytes) ----------------
constexpr size_t SZ_XBF = (size_t)M * D * 2;
constexpr size_t SZ_W   = (size_t)D * D * 2;
constexpr size_t SZ_Q   = (size_t)NB * NH * S * HD * 2;
constexpr size_t SZ_KP  = (size_t)NB * NH * SP * HD * 2;
constexpr size_t OFF_XBF = 0;
constexpr size_t OFF_WQT = OFF_XBF + SZ_XBF;
constexpr size_t OFF_WKT = OFF_WQT + SZ_W;
constexpr size_t OFF_WVT = OFF_WKT + SZ_W;
constexpr size_t OFF_WOT = OFF_WVT + SZ_W;
constexpr size_t OFF_Q   = OFF_WOT + SZ_W;
constexpr size_t OFF_KP  = OFF_Q + SZ_Q;
constexpr size_t OFF_VT  = OFF_KP + SZ_KP;
constexpr size_t OFF_AO  = OFF_VT + SZ_KP;

// ---------------- conversion kernels ----------------
__global__ __launch_bounds__(256) void cvt_f32_bf16(const float* __restrict__ in,
                                                    bf16_t* __restrict__ out, int n) {
  for (int i = blockIdx.x * blockDim.x + threadIdx.x; i < n; i += gridDim.x * blockDim.x)
    out[i] = (bf16_t)in[i];
}

// WT[n][k] = W[k][n]  (1024x1024)
__global__ __launch_bounds__(256) void cvt_transpose_w(const float* __restrict__ W,
                                                       bf16_t* __restrict__ WT) {
  int idx = blockIdx.x * 256 + threadIdx.x;   // 1M threads
  int n = idx & (D - 1);
  int k = idx >> 10;
  WT[(size_t)n * D + k] = (bf16_t)W[(size_t)k * D + n];
}

// ---------------- shared GEMM core: C[128x128] += A[128xK] * BT[128xK]^T ----------------
// A: [m][k] bf16 row-major, BT: [n][k] bf16 row-major. Double-buffered LDS tiles,
// filled with async global->LDS copies one K-step ahead (ASYNCcnt pipelined).
// 256 threads = 8 waves; wave grid 4(M) x 2(N); each wave: 2x4 sub-tiles of 16x16.
__device__ __forceinline__ void gemm_core(const bf16_t* __restrict__ A,
                                          const bf16_t* __restrict__ BT,
                                          int M0, int N0, int K, int lda, int ldb,
                                          bf16_t* Alds, bf16_t* Blds,
                                          v8f acc[2][4]) {
  const int tid  = threadIdx.x;
  const int lane = tid & 31;
  const int wid  = tid >> 5;
  const int wm   = wid & 3;
  const int wn   = wid >> 2;
  const int hf   = lane >> 4;
  const int l15  = lane & 15;
  const int lrow = tid >> 1;           // 0..127
  const int lchk = (tid & 1) * 16;     // 0 or 16 bf16

  const bf16_t* gA = A  + (size_t)(M0 + lrow) * lda + lchk;
  const bf16_t* gB = BT + (size_t)(N0 + lrow) * ldb + lchk;
  bf16_t* lA = Alds + lrow * 32 + lchk;
  bf16_t* lB = Blds + lrow * 32 + lchk;

  // prologue: stage K-tile 0 into buffer 0
  stage_tile32(gA, lA);
  stage_tile32(gB, lB);

  for (int k0 = 0; k0 < K; k0 += 32) {
    const int cur = (k0 >> 5) & 1;
    const int nxt = cur ^ 1;
    if (k0 + 32 < K) {
      // stage next K-tile into the other buffer (4 async ops)
      stage_tile32(gA + k0 + 32, lA + nxt * 4096);
      stage_tile32(gB + k0 + 32, lB + nxt * 4096);
      ASYNC_WAIT(4);   // ASYNC ops retire in order: current tile (older 4) complete
    } else {
      ASYNC_WAIT(0);
    }
    __syncthreads();   // every wave's portion of the current tile is in LDS

    const bf16_t* Ab = Alds + cur * 4096;
    const bf16_t* Bb = Blds + cur * 4096;
    Frag16 afr[2], bfr[4];
#pragma unroll
    for (int mi = 0; mi < 2; ++mi) {
      const v8bf* Ar = (const v8bf*)(Ab + (wm * 32 + mi * 16 + l15) * 32);
      afr[mi].h[0] = Ar[hf];        // K = 8*hf .. 8*hf+7
      afr[mi].h[1] = Ar[2 + hf];    // K = 16+8*hf .. +7
    }
#pragma unroll
    for (int ni = 0; ni < 4; ++ni) {
      const v8bf* Br = (const v8bf*)(Bb + (wn * 64 + ni * 16 + l15) * 32);
      bfr[ni].h[0] = Br[hf];            // K = 8*hf .. 8*hf+7
      bfr[ni].h[1] = Br[2 + hf];        // K = 16+8*hf .. +7
    }
#pragma unroll
    for (int mi = 0; mi < 2; ++mi)
#pragma unroll
      for (int ni = 0; ni < 4; ++ni)
        acc[mi][ni] = WMMA_BF16(afr[mi].v, bfr[ni].v, acc[mi][ni]);
    __syncthreads();   // all waves done reading before buffer is overwritten
  }
}

// ---------------- QKV projection ----------------
// grid (M/128, D/128, 3): z selects Q/K/V. Scatters into attention-friendly layouts.
__global__ __launch_bounds__(256) void qkv_gemm(const bf16_t* __restrict__ xbf,
                                                const bf16_t* __restrict__ WqT,
                                                const bf16_t* __restrict__ WkT,
                                                const bf16_t* __restrict__ WvT,
                                                const float* __restrict__ bq,
                                                const float* __restrict__ bk,
                                                const float* __restrict__ bv,
                                                bf16_t* __restrict__ Q,
                                                bf16_t* __restrict__ Kp,
                                                bf16_t* __restrict__ Vt) {
  __shared__ bf16_t Alds[2 * 128 * 32];
  __shared__ bf16_t Blds[2 * 128 * 32];
  const int which = blockIdx.z;
  const bf16_t* BT  = (which == 0) ? WqT : (which == 1) ? WkT : WvT;
  const float* bias = (which == 0) ? bq  : (which == 1) ? bk  : bv;
  const int M0 = blockIdx.x * 128;
  const int N0 = blockIdx.y * 128;

  v8f acc[2][4] = {};
  gemm_core(xbf, BT, M0, N0, D, D, D, Alds, Blds, acc);

  const int lane = threadIdx.x & 31;
  const int wid  = threadIdx.x >> 5;
  const int wm   = wid & 3;
  const int wn   = wid >> 2;
  const int hf   = lane >> 4;
  const int l15  = lane & 15;

#pragma unroll
  for (int mi = 0; mi < 2; ++mi) {
#pragma unroll
    for (int ni = 0; ni < 4; ++ni) {
      const int n = N0 + wn * 64 + ni * 16 + l15;
      const int h = n >> 6, dd = n & 63;
      const float bn = bias[n];
#pragma unroll
      for (int v = 0; v < 8; ++v) {
        const int m = M0 + wm * 32 + mi * 16 + v + 8 * hf;
        const int b = m >> 12, s = m & (S - 1);
        const float c = acc[mi][ni][v] + bn;
        const size_t bh = (size_t)b * NH + h;
        if (which == 0) {
          Q[(bh * S + s) * HD + dd] = (bf16_t)(c * 0.125f);   // fold 1/sqrt(64)
        } else if (which == 1) {
          Kp[(bh * SP + (s + C)) * HD + dd] = (bf16_t)c;
        } else {
          Vt[(bh * HD + dd) * SP + (s + C)] = (bf16_t)c;      // transposed V
        }
      }
    }
  }
}

// ---------------- sliding-window attention (flash-style) ----------------
// grid (S/C=32, NH, NB); 8 waves, each wave owns a 16-row q strip.
__global__ __launch_bounds__(256) void attn_kernel(const bf16_t* __restrict__ Q,
                                                   const bf16_t* __restrict__ Kp,
                                                   const bf16_t* __restrict__ Vt,
                                                   const float* __restrict__ rpb,
                                                   bf16_t* __restrict__ attn_out) {
  const int qb = blockIdx.x;   // q block 0..31
  const int h  = blockIdx.y;
  const int b  = blockIdx.z;

  __shared__ float  bias_lds[2 * C + 1];
  __shared__ bf16_t Pstage[8][16 * 32];

  const int tid = threadIdx.x;
  for (int i = tid; i < 2 * C + 1; i += 256) bias_lds[i] = rpb[h * (2 * C + 1) + i];
  __syncthreads();

  const int wave = tid >> 5;
  const int lane = tid & 31;
  const int hf   = lane >> 4;
  const int l15  = lane & 15;
  const size_t bh = (size_t)b * NH + h;

  const bf16_t* Qbase = Q  + (bh * S + (size_t)qb * C + wave * 16) * HD;
  const bf16_t* Kbase = Kp + (bh * SP + (size_t)qb * C) * HD;   // padded window start
  const bf16_t* Vbase = Vt + bh * HD * SP + (size_t)qb * C;

  // Q A-fragments for d=0..31 and d=32..63, reused across the whole window.
  Frag16 qf[2];
#pragma unroll
  for (int kk = 0; kk < 2; ++kk) {
    const v8bf* Qr = (const v8bf*)(Qbase + l15 * HD + kk * 32);
    qf[kk].h[0] = Qr[hf];
    qf[kk].h[1] = Qr[2 + hf];
  }

  float m_run[8], l_run[8];
  v8f oacc[4] = {};
#pragma unroll
  for (int v = 0; v < 8; ++v) { m_run[v] = -1e30f; l_run[v] = 0.f; }

  bf16_t* Pst = &Pstage[wave][0];

  for (int pt = 0; pt < WIN / 32; ++pt) {   // 12 pairs of 16-col score tiles
    v8f sc[2];
#pragma unroll
    for (int t = 0; t < 2; ++t) {
      v8f a = {};
      const int ncol = pt * 32 + t * 16;
#pragma unroll
      for (int kk = 0; kk < 2; ++kk) {
        Frag16 kf;
        const v8bf* Kr = (const v8bf*)(Kbase + (size_t)(ncol + l15) * HD + kk * 32);
        kf.h[0] = Kr[hf];
        kf.h[1] = Kr[2 + hf];
        a = WMMA_BF16(qf[kk].v, kf.v, a);
      }
      sc[t] = a;
    }

    // bias + band/key mask, per-row max over this 32-wide slab
    float sv[2][8], pm[8];
#pragma unroll
    for (int v = 0; v < 8; ++v) pm[v] = -1e30f;
#pragma unroll
    for (int t = 0; t < 2; ++t) {
      const int w = pt * 32 + t * 16 + l15;
      const int kglob = qb * C - C + w;
      const bool keyok = (kglob >= 0) && (kglob < S);
#pragma unroll
      for (int v = 0; v < 8; ++v) {
        const int qpos = wave * 16 + v + 8 * hf;
        const int rel = w - qpos;
        const int ridx = rel < 0 ? 0 : (rel > 2 * C ? 2 * C : rel);
        const bool ok = keyok && (rel >= 0) && (rel <= 2 * C);
        const float val = ok ? (sc[t][v] + bias_lds[ridx]) : -1e30f;
        sv[t][v] = val;
        pm[v] = fmaxf(pm[v], val);
      }
    }
#pragma unroll
    for (int v = 0; v < 8; ++v) {   // reduce across the 16 lanes of this half
      float x = pm[v];
      x = fmaxf(x, __shfl_xor(x, 1, 32));
      x = fmaxf(x, __shfl_xor(x, 2, 32));
      x = fmaxf(x, __shfl_xor(x, 4, 32));
      x = fmaxf(x, __shfl_xor(x, 8, 32));
      pm[v] = x;
    }

    float rescv[8], psum[8];
#pragma unroll
    for (int v = 0; v < 8; ++v) {
      const float mnew = fmaxf(m_run[v], pm[v]);
      const float resc = __expf(m_run[v] - mnew);
      m_run[v] = mnew;
      rescv[v] = resc;
      const float p0 = __expf(sv[0][v] - mnew);
      const float p1 = __expf(sv[1][v] - mnew);
      psum[v] = p0 + p1;
      l_run[v] *= resc;
      Pst[(v + 8 * hf) * 32 + l15]      = (bf16_t)p0;  // relayout C-frag -> A-frag
      Pst[(v + 8 * hf) * 32 + 16 + l15] = (bf16_t)p1;
    }
#pragma unroll
    for (int v = 0; v < 8; ++v) {
      float x = psum[v];
      x += __shfl_xor(x, 1, 32);
      x += __shfl_xor(x, 2, 32);
      x += __shfl_xor(x, 4, 32);
      x += __shfl_xor(x, 8, 32);
      l_run[v] += x;
    }
#pragma unroll
    for (int ni = 0; ni < 4; ++ni)
#pragma unroll
      for (int v = 0; v < 8; ++v) oacc[ni][v] *= rescv[v];

    // P @ V for this 32-wide slab
    Frag16 pf;
    {
      const v8bf* Pr = (const v8bf*)(Pst + l15 * 32);
      pf.h[0] = Pr[hf];
      pf.h[1] = Pr[2 + hf];
    }
#pragma unroll
    for (int ni = 0; ni < 4; ++ni) {
      Frag16 vf;
      const v8bf* Vr = (const v8bf*)(Vbase + (size_t)(ni * 16 + l15) * SP + pt * 32);
      vf.h[0] = Vr[hf];
      vf.h[1] = Vr[2 + hf];
      oacc[ni] = WMMA_BF16(pf.v, vf.v, oacc[ni]);
    }
  }

  // normalize and store head-concatenated rows
#pragma unroll
  for (int ni = 0; ni < 4; ++ni) {
#pragma unroll
    for (int v = 0; v < 8; ++v) {
      const float val = oacc[ni][v] / l_run[v];
      const int srow = qb * C + wave * 16 + v + 8 * hf;
      const int col  = h * HD + ni * 16 + l15;
      attn_out[((size_t)b * S + srow) * D + col] = (bf16_t)val;
    }
  }
}

// ---------------- output projection ----------------
__global__ __launch_bounds__(256) void out_gemm(const bf16_t* __restrict__ Abf,
                                                const bf16_t* __restrict__ WoT,
                                                const float* __restrict__ bo,
                                                float* __restrict__ out) {
  __shared__ bf16_t Alds[2 * 128 * 32];
  __shared__ bf16_t Blds[2 * 128 * 32];
  const int M0 = blockIdx.x * 128;
  const int N0 = blockIdx.y * 128;

  v8f acc[2][4] = {};
  gemm_core(Abf, WoT, M0, N0, D, D, D, Alds, Blds, acc);

  const int lane = threadIdx.x & 31;
  const int wid  = threadIdx.x >> 5;
  const int wm   = wid & 3;
  const int wn   = wid >> 2;
  const int hf   = lane >> 4;
  const int l15  = lane & 15;

#pragma unroll
  for (int mi = 0; mi < 2; ++mi) {
#pragma unroll
    for (int ni = 0; ni < 4; ++ni) {
      const int n = N0 + wn * 64 + ni * 16 + l15;
      const float bn = bo[n];
#pragma unroll
      for (int v = 0; v < 8; ++v) {
        const int m = M0 + wm * 32 + mi * 16 + v + 8 * hf;
        out[(size_t)m * D + n] = acc[mi][ni][v] + bn;
      }
    }
  }
}

// ---------------- host launcher ----------------
extern "C" void kernel_launch(void* const* d_in, const int* in_sizes, int n_in,
                              void* d_out, int out_size, void* d_ws, size_t ws_size,
                              hipStream_t stream) {
  const float* x   = (const float*)d_in[0];
  const float* Wq  = (const float*)d_in[1];
  const float* bq  = (const float*)d_in[2];
  const float* Wk  = (const float*)d_in[3];
  const float* bk  = (const float*)d_in[4];
  const float* Wv  = (const float*)d_in[5];
  const float* bv  = (const float*)d_in[6];
  const float* Wo  = (const float*)d_in[7];
  const float* bo  = (const float*)d_in[8];
  const float* rpb = (const float*)d_in[9];

  char* ws = (char*)d_ws;
  bf16_t* xbf = (bf16_t*)(ws + OFF_XBF);
  bf16_t* WqT = (bf16_t*)(ws + OFF_WQT);
  bf16_t* WkT = (bf16_t*)(ws + OFF_WKT);
  bf16_t* WvT = (bf16_t*)(ws + OFF_WVT);
  bf16_t* WoT = (bf16_t*)(ws + OFF_WOT);
  bf16_t* Qb  = (bf16_t*)(ws + OFF_Q);
  bf16_t* Kp  = (bf16_t*)(ws + OFF_KP);
  bf16_t* Vt  = (bf16_t*)(ws + OFF_VT);
  bf16_t* Ao  = (bf16_t*)(ws + OFF_AO);

  // zero K/V padding rows (window edges read them; masked, but keep them finite)
  (void)hipMemsetAsync(Kp, 0, SZ_KP, stream);
  (void)hipMemsetAsync(Vt, 0, SZ_KP, stream);

  cvt_f32_bf16<<<8192, 256, 0, stream>>>(x, xbf, M * D);
  cvt_transpose_w<<<(D * D) / 256, 256, 0, stream>>>(Wq, WqT);
  cvt_transpose_w<<<(D * D) / 256, 256, 0, stream>>>(Wk, WkT);
  cvt_transpose_w<<<(D * D) / 256, 256, 0, stream>>>(Wv, WvT);
  cvt_transpose_w<<<(D * D) / 256, 256, 0, stream>>>(Wo, WoT);

  qkv_gemm<<<dim3(M / 128, D / 128, 3), 256, 0, stream>>>(xbf, WqT, WkT, WvT,
                                                          bq, bk, bv, Qb, Kp, Vt);

  attn_kernel<<<dim3(S / C, NH, NB), 256, 0, stream>>>(Qb, Kp, Vt, rpb, Ao);

  out_gemm<<<dim3(M / 128, D / 128), 256, 0, stream>>>(Ao, WoT, bo, (float*)d_out);
}
